// DKN_28106265985609
// MI455X (gfx1250) — compile-verified, matrix operand order checked by reading, not run on hardware
//
#include <hip/hip_runtime.h>
#include <hip/hip_bf16.h>

typedef __attribute__((ext_vector_type(16))) _Float16 v16h;
typedef __attribute__((ext_vector_type(8)))  _Float16 v8h;
typedef __attribute__((ext_vector_type(8)))  float    v8f;

#define NSHIFT 16
#define PADH   303
#define KMAX   1600   // largest Kpad (c5: 64*5*5), already a multiple of 64

// ---------------------------------------------------------------------------
// Pad + f16 convert: build (16, Cin, 303, 303) shifted copies of a (Cin,256,256)
// fp32 source. Copy s = i*4+j has top-pad (25-i), left-pad (25-j).
// ---------------------------------------------------------------------------
__global__ void dkn_pad_input(const float* __restrict__ src, _Float16* __restrict__ dst,
                              int Cin) {
  long long idx = (long long)blockIdx.x * blockDim.x + threadIdx.x;
  long long total = (long long)NSHIFT * Cin * PADH * PADH;
  if (idx >= total) return;
  int x = (int)(idx % PADH);
  long long t = idx / PADH;
  int y = (int)(t % PADH); t /= PADH;
  int c = (int)(t % Cin);
  int s = (int)(t / Cin);
  int i = s >> 2, j = s & 3;
  int sy = y - (25 - i);
  int sx = x - (25 - j);
  float v = 0.f;
  if (sy >= 0 && sy < 256 && sx >= 0 && sx < 256)
    v = src[(c * 256 + sy) * 256 + sx];
  dst[idx] = (_Float16)v;
}

// ---------------------------------------------------------------------------
// Pack fp32 OIHW weights -> f16 [CoutPad][Kpad] GEMM A-matrix (zero padded).
// OIHW flat index = co*Kreal + k with k = (ci*KH+kh)*KW+kw.
// ---------------------------------------------------------------------------
__global__ void dkn_pack_weights(const float* __restrict__ w, _Float16* __restrict__ wf,
                                 int Cout, int Kreal, int CoutPad, int Kpad) {
  int idx = blockIdx.x * blockDim.x + threadIdx.x;
  int total = CoutPad * Kpad;
  if (idx >= total) return;
  int co = idx / Kpad;
  int k  = idx - co * Kpad;
  float v = 0.f;
  if (co < Cout && k < Kreal) v = w[(size_t)co * Kreal + k];
  wf[idx] = (_Float16)v;
}

// ---------------------------------------------------------------------------
// Implicit-GEMM convolution with V_WMMA_F32_16X16X32_F16.
//   Block tile: N = 32 pixels, K-chunk = 64, M = CoutPad (16 rows per wave).
//   Block = 2*CoutPad threads = CoutPad/16 waves; 4 WMMAs per wave per chunk.
//   Double-buffered LDS B tile: gather for chunk c+1 overlaps WMMA of chunk c.
// act: 0 = bias+relu, 1 = bias+bn+relu, 2 = bias+sigmoid, 3 = bias only.
// ---------------------------------------------------------------------------
struct ConvArgs {
  const _Float16* __restrict__ in;     // (16, Cin, IH, IW) f16
  const _Float16* __restrict__ wf;     // (CoutPad, Kpad)   f16
  const float* __restrict__ bias;
  const float* __restrict__ bng;
  const float* __restrict__ bnb;
  const float* __restrict__ bnm;
  const float* __restrict__ bnv;
  _Float16* __restrict__ outh;         // f16 NCHW out (nullable)
  float*    __restrict__ outf;         // f32 NCHW out (nullable)
  int Cin, Cout, CoutPad, KH, KW, stride;
  int IH, IW, OH, OW;
  int Kreal, Kpad, act;
};

__global__ void dkn_conv_wmma(ConvArgs a) {
  // B tiles: [buf][pixel 0..31][k 0..63]; rows padded to 72 halves (144 B) so
  // 16-lane column reads (stride 36 banks) are conflict-free.
  __shared__ __align__(16) _Float16 Bt[2][32][72];
  __shared__ int ktab[KMAX];     // k -> (ci*IH+kh)*IW+kw, or -1 for K padding

  const int P       = NSHIFT * a.OH * a.OW;
  const int pixBase = blockIdx.x * 32;
  const int tid  = threadIdx.x;
  const int lane = tid & 31;
  const int wave = tid >> 5;
  const int nn   = lane & 15;   // A row / B column / D column (within tile)
  const int hf   = lane >> 4;   // lane half selects K sub-block
  const int KHW  = a.KH * a.KW;
  const int OHW  = a.OH * a.OW;

  // Build the k -> input-offset table once (divisions amortized over all chunks).
  for (int k = tid; k < a.Kpad; k += blockDim.x) {
    int t = -1;
    if (k < a.Kreal) {
      int ci = k / KHW;
      int r  = k - ci * KHW;
      int kh = r / a.KW;
      int kw = r - kh * a.KW;
      t = (ci * a.IH + kh) * a.IW + kw;
    }
    ktab[k] = t;
  }

  // Per-thread loader pixel: fixed column p = tid&31 across the whole K loop.
  const int lp  = tid & 31;
  const int lpg = pixBase + lp;
  int  pbOff  = 0;
  bool pvalid = (lpg < P);
  if (pvalid) {
    int n  = lpg / OHW;
    int r2 = lpg - n * OHW;
    int oy = r2 / a.OW;
    int ox = r2 - oy * a.OW;
    pbOff = ((n * a.Cin) * a.IH + oy * a.stride) * a.IW + ox * a.stride;
  }

  const _Float16* wrow = a.wf + (size_t)(wave * 16 + nn) * a.Kpad;
  __builtin_prefetch(wrow, 0, 3);

  v8f acc0 = {0.f, 0.f, 0.f, 0.f, 0.f, 0.f, 0.f, 0.f};
  v8f acc1 = {0.f, 0.f, 0.f, 0.f, 0.f, 0.f, 0.f, 0.f};

  const int kkStep = blockDim.x >> 5;   // k-rows advanced per loader pass
  auto load_tile = [&](int kc, int buf) {
    for (int kk = tid >> 5; kk < 64; kk += kkStep) {
      int t = ktab[kc + kk];
      _Float16 v = (_Float16)0.f;
      if (t >= 0 && pvalid) v = a.in[pbOff + t];
      Bt[buf][lp][kk] = v;
    }
  };

  const int nChunks = a.Kpad >> 6;
  __syncthreads();            // ktab visible to all loader threads
  load_tile(0, 0);

  for (int c = 0; c < nChunks; ++c) {
    __syncthreads();          // tile c fully staged; previous compute done
    if (c + 1 < nChunks) load_tile((c + 1) << 6, (c + 1) & 1);  // prefetch next
    const int kc  = c << 6;
    const int buf = c & 1;

    // A fragments (row nn of packed weights), K sub-blocks [kc,kc+32), [kc+32,kc+64)
    v8h a0 = *(const v8h*)(wrow + kc + hf * 8);
    v8h a1 = *(const v8h*)(wrow + kc + 16 + hf * 8);
    v8h a2 = *(const v8h*)(wrow + kc + 32 + hf * 8);
    v8h a3 = *(const v8h*)(wrow + kc + 48 + hf * 8);
    v16h A0 = __builtin_shufflevector(a0, a1, 0,1,2,3,4,5,6,7,8,9,10,11,12,13,14,15);
    v16h A1 = __builtin_shufflevector(a2, a3, 0,1,2,3,4,5,6,7,8,9,10,11,12,13,14,15);

    // B fragments: pixel columns [0,16) and [16,32), same K mapping as A.
    v8h b00 = *(const v8h*)(&Bt[buf][nn][hf * 8]);
    v8h b01 = *(const v8h*)(&Bt[buf][nn][16 + hf * 8]);
    v8h b02 = *(const v8h*)(&Bt[buf][nn][32 + hf * 8]);
    v8h b03 = *(const v8h*)(&Bt[buf][nn][48 + hf * 8]);
    v8h b10 = *(const v8h*)(&Bt[buf][nn + 16][hf * 8]);
    v8h b11 = *(const v8h*)(&Bt[buf][nn + 16][16 + hf * 8]);
    v8h b12 = *(const v8h*)(&Bt[buf][nn + 16][32 + hf * 8]);
    v8h b13 = *(const v8h*)(&Bt[buf][nn + 16][48 + hf * 8]);
    v16h B00 = __builtin_shufflevector(b00, b01, 0,1,2,3,4,5,6,7,8,9,10,11,12,13,14,15);
    v16h B01 = __builtin_shufflevector(b02, b03, 0,1,2,3,4,5,6,7,8,9,10,11,12,13,14,15);
    v16h B10 = __builtin_shufflevector(b10, b11, 0,1,2,3,4,5,6,7,8,9,10,11,12,13,14,15);
    v16h B11 = __builtin_shufflevector(b12, b13, 0,1,2,3,4,5,6,7,8,9,10,11,12,13,14,15);

    acc0 = __builtin_amdgcn_wmma_f32_16x16x32_f16(false, A0, false, B00, (short)0, acc0, false, false);
    acc1 = __builtin_amdgcn_wmma_f32_16x16x32_f16(false, A0, false, B10, (short)0, acc1, false, false);
    acc0 = __builtin_amdgcn_wmma_f32_16x16x32_f16(false, A1, false, B01, (short)0, acc0, false, false);
    acc1 = __builtin_amdgcn_wmma_f32_16x16x32_f16(false, A1, false, B11, (short)0, acc1, false, false);
  }

  // Epilogue. D layout: lane<16 -> (M=vr, N=lane); lane>=16 -> (M=8+vr, N=lane-16).
  const int mbase = hf * 8;
  for (int half = 0; half < 2; ++half) {
    int pg = pixBase + half * 16 + nn;
    if (pg >= P) continue;
    int n  = pg / OHW;
    int r2 = pg - n * OHW;
    int oy = r2 / a.OW;
    int ox = r2 - oy * a.OW;
    const v8f& acc = half ? acc1 : acc0;
    for (int vr = 0; vr < 8; ++vr) {
      int co = wave * 16 + mbase + vr;
      if (co >= a.Cout) continue;
      float v = acc[vr] + a.bias[co];
      if (a.act == 1) {
        float sc = a.bng[co] * rsqrtf(a.bnv[co] + 1e-5f);
        v = (v - a.bnm[co]) * sc + a.bnb[co];
        v = fmaxf(v, 0.f);
      } else if (a.act == 0) {
        v = fmaxf(v, 0.f);
      } else if (a.act == 2) {
        v = 1.f / (1.f + __expf(-v));
      }
      size_t oidx = (((size_t)n * a.Cout + co) * a.OH + oy) * a.OW + ox;
      if (a.outh) a.outh[oidx] = (_Float16)v;
      if (a.outf) a.outf[oidx] = v;
    }
  }
}

// ---------------------------------------------------------------------------
// Fused combine + stitch + deformable bilinear sampling + residual.
// wim/wd: (16,9,64,64) f32, oim/od: (16,18,64,64) f32.
// For output pixel (y,x): s=(y%4)*4+(x%4), tile (yq,xq)=(y/4,x/4).
// Sample coord within 15x15 patch: (fx,fy) = (6.5+a+offx, 6.5+b+offy);
// patch(ky,kx) = lr[y+ky-7, x+kx-7] (zero outside image / outside [0,14]).
// ---------------------------------------------------------------------------
__global__ void dkn_finalize(const float* __restrict__ lr,
                             const float* __restrict__ wim, const float* __restrict__ wd,
                             const float* __restrict__ oim, const float* __restrict__ od,
                             float* __restrict__ out) {
  int idx = blockIdx.x * blockDim.x + threadIdx.x;
  if (idx >= 256 * 256) return;
  int x = idx & 255, y = idx >> 8;
  int s  = (y & 3) * 4 + (x & 3);
  int yq = y >> 2,  xq = x >> 2;
  int pbase = yq * 64 + xq;

  float w9[9], mean = 0.f;
  for (int r = 0; r < 9; ++r) {
    size_t ii = (size_t)(s * 9 + r) * 4096 + pbase;
    float wv = wim[ii] * wd[ii];
    w9[r] = wv; mean += wv;
  }
  mean *= (1.f / 9.f);

  float accv = 0.f;
  for (int aa = 0; aa < 3; ++aa) {
    for (int bb = 0; bb < 3; ++bb) {
      int c0 = aa * 6 + bb * 2;
      size_t i0 = (size_t)(s * 18 + c0)     * 4096 + pbase;
      size_t i1 = (size_t)(s * 18 + c0 + 1) * 4096 + pbase;
      float offx = oim[i0] * od[i0];
      float offy = oim[i1] * od[i1];
      float fx = 6.5f + aa + offx;
      float fy = 6.5f + bb + offy;
      float x0 = floorf(fx), y0 = floorf(fy);
      float wx1 = fx - x0,   wy1 = fy - y0;
      float sv = 0.f;
      for (int dy = 0; dy < 2; ++dy) {
        for (int dx = 0; dx < 2; ++dx) {
          float xx = x0 + dx, yy = y0 + dy;
          float wgt = (dx ? wx1 : 1.f - wx1) * (dy ? wy1 : 1.f - wy1);
          float v = 0.f;
          if (xx >= 0.f && xx <= 14.f && yy >= 0.f && yy <= 14.f) {
            int gx = x + (int)xx - 7;
            int gy = y + (int)yy - 7;
            if (gx >= 0 && gx < 256 && gy >= 0 && gy < 256) v = lr[gy * 256 + gx];
          }
          sv += v * wgt;
        }
      }
      accv += sv * (w9[aa * 3 + bb] - mean);
    }
  }
  out[idx] = accv + lr[idx];
}

// ---------------------------------------------------------------------------
// Host orchestration.
// Input layout: d_in[0]=image, d_in[1]=lr, then 30 leaves per tower dict in
// sorted-key order (im_params first, then d_params).
// ---------------------------------------------------------------------------
enum {
  BN1_B = 0, BN1_G, BN1_M, BN1_V,
  BN3_B, BN3_G, BN3_M, BN3_V,
  BN5_B, BN5_G, BN5_M, BN5_V,
  C1_B, C1_W, C2_B, C2_W, C3_B, C3_W, C4_B, C4_W,
  C5_B, C5_W, C6_B, C6_W, C7_B, C7_W,
  CO_B, CO_W, CW_B, CW_W, TOWER_N
};

extern "C" void kernel_launch(void* const* d_in, const int* in_sizes, int n_in,
                              void* d_out, int out_size, void* d_ws, size_t ws_size,
                              hipStream_t stream) {
  (void)in_sizes; (void)n_in; (void)out_size; (void)ws_size;
  const float* image = (const float*)d_in[0];
  const float* lr    = (const float*)d_in[1];
  auto TP = [&](int t, int k) -> const float* {
    return (const float*)d_in[2 + t * TOWER_N + k];
  };

  char* ws = (char*)d_ws;
  size_t off = 0;
  auto alloc = [&](size_t bytes) -> void* {
    off = (off + 255) & ~(size_t)255;
    void* p = ws + off;
    off += bytes;
    return p;
  };

  _Float16* xin   = (_Float16*)alloc((size_t)NSHIFT * 3 * PADH * PADH * 2);
  _Float16* bufA  = (_Float16*)alloc((size_t)NSHIFT * 32 * 297 * 297 * 2);
  _Float16* bufB  = (_Float16*)alloc((size_t)NSHIFT * 32 * 148 * 148 * 2);
  _Float16* wfbuf = (_Float16*)alloc((size_t)128 * KMAX * 2);
  float* wout[2] = { (float*)alloc((size_t)NSHIFT * 9  * 64 * 64 * 4),
                     (float*)alloc((size_t)NSHIFT * 9  * 64 * 64 * 4) };
  float* oout[2] = { (float*)alloc((size_t)NSHIFT * 18 * 64 * 64 * 4),
                     (float*)alloc((size_t)NSHIFT * 18 * 64 * 64 * 4) };

  struct LDef { int wI, bI, gI, bbI, mI, vI, Cout, KH, KW, stride, act; };
  static const LDef L[9] = {
    {C1_W, C1_B, BN1_G, BN1_B, BN1_M, BN1_V,  32, 7, 7, 1, 1},
    {C2_W, C2_B, -1, -1, -1, -1,              32, 2, 2, 2, 0},
    {C3_W, C3_B, BN3_G, BN3_B, BN3_M, BN3_V,  64, 5, 5, 1, 1},
    {C4_W, C4_B, -1, -1, -1, -1,              64, 2, 2, 2, 0},
    {C5_W, C5_B, BN5_G, BN5_B, BN5_M, BN5_V, 128, 5, 5, 1, 1},
    {C6_W, C6_B, -1, -1, -1, -1,             128, 3, 3, 1, 0},
    {C7_W, C7_B, -1, -1, -1, -1,             128, 3, 3, 1, 0},
    {CW_W, CW_B, -1, -1, -1, -1,               9, 1, 1, 1, 2},
    {CO_W, CO_B, -1, -1, -1, -1,              18, 1, 1, 1, 3},
  };

  for (int t = 0; t < 2; ++t) {
    const int Cin0 = (t == 0) ? 3 : 1;
    const float* src = (t == 0) ? image : lr;
    {
      long long total = (long long)NSHIFT * Cin0 * PADH * PADH;
      dkn_pad_input<<<(int)((total + 255) / 256), 256, 0, stream>>>(src, xin, Cin0);
    }

    const _Float16* cur = xin;
    int IH = PADH, IW = PADH, Cin = Cin0;
    for (int li = 0; li < 9; ++li) {
      const LDef& d = L[li];
      int OH = (IH - d.KH) / d.stride + 1;
      int OW = (IW - d.KW) / d.stride + 1;
      int Kreal   = Cin * d.KH * d.KW;
      int Kpad    = (Kreal + 63) & ~63;   // K-chunk is 64
      int CoutPad = (d.Cout + 15) & ~15;

      {
        int total = CoutPad * Kpad;
        dkn_pack_weights<<<(total + 255) / 256, 256, 0, stream>>>(
            TP(t, d.wI), wfbuf, d.Cout, Kreal, CoutPad, Kpad);
      }

      ConvArgs ca = {};
      ca.in = cur; ca.wf = wfbuf; ca.bias = TP(t, d.bI);
      if (d.gI >= 0) {
        ca.bng = TP(t, d.gI); ca.bnb = TP(t, d.bbI);
        ca.bnm = TP(t, d.mI); ca.bnv = TP(t, d.vI);
      }
      ca.Cin = Cin; ca.Cout = d.Cout; ca.CoutPad = CoutPad;
      ca.KH = d.KH; ca.KW = d.KW; ca.stride = d.stride;
      ca.IH = IH; ca.IW = IW; ca.OH = OH; ca.OW = OW;
      ca.Kreal = Kreal; ca.Kpad = Kpad; ca.act = d.act;

      _Float16* dsth = nullptr;
      if (li < 7)      { dsth = (li % 2 == 0) ? bufA : bufB; ca.outh = dsth; }
      else if (li == 7) ca.outf = wout[t];
      else              ca.outf = oout[t];

      int P = NSHIFT * OH * OW;
      dkn_conv_wmma<<<dim3((P + 31) / 32), 2 * CoutPad, 0, stream>>>(ca);

      if (li < 7) { cur = dsth; Cin = d.Cout; IH = OH; IW = OW; }
      // li==7/8 both consume the layer-6 activations (cur stays bufA, 128ch, 64x64)
    }
  }

  dkn_finalize<<<(65536 + 255) / 256, 256, 0, stream>>>(
      lr, wout[0], wout[1], oout[0], oout[1], (float*)d_out);
}
